// LocalTransformerLayer_47605417509009
// MI455X (gfx1250) — compile-verified, matrix operand order checked by reading, not run on hardware
//
#include <hip/hip_runtime.h>
#include <hip/hip_bf16.h>

#define N_NODES 16384
#define C_DIM   512
#define E_EDGES 524288
#define B_GR    64
#define NPG     256
#define H_HEADS 8
#define EF_DIM  16

typedef __attribute__((ext_vector_type(16))) __bf16 bf16x16;
typedef __attribute__((ext_vector_type(8)))  __bf16 bf16x8;
typedef __attribute__((ext_vector_type(8)))  float  f32x8;

#if defined(__AMDGCN__) && __has_builtin(__builtin_amdgcn_global_load_async_to_lds_b128) && \
    __has_builtin(__builtin_amdgcn_s_wait_asynccnt)
#define USE_ASYNC_LDS 1
typedef int v4i __attribute__((vector_size(16)));
typedef __attribute__((address_space(1))) v4i as1_v4i;
typedef __attribute__((address_space(3))) v4i as3_v4i;
#else
#define USE_ASYNC_LDS 0
#endif

__device__ __forceinline__ f32x8 wmma_bf16(bf16x16 a, bf16x16 b, f32x8 c) {
  return __builtin_amdgcn_wmma_f32_16x16x32_bf16(false, a, false, b, (short)0, c, false, false);
}

__device__ __forceinline__ float wave_sum32(float v) {
#pragma unroll
  for (int m = 16; m >= 1; m >>= 1) v += __shfl_xor(v, m, 32);
  return v;
}
__device__ __forceinline__ float wave_max32(float v) {
#pragma unroll
  for (int m = 16; m >= 1; m >>= 1) v = fmaxf(v, __shfl_xor(v, m, 32));
  return v;
}

// ---------------------------------------------------------------------------
// Pack fp32 weight (K x Nc, row-major) into WMMA B-fragment-native bf16 layout:
// tile = ntile*KT + ktile ; per tile, lane L stores its 16 B-fragment bf16s
// contiguously: lanes 0-15 column n=nt*16+L with K = kt*32+{0..7,16..23},
// lanes 16-31 same column with K = kt*32+{8..15,24..31}.
// ---------------------------------------------------------------------------
__global__ void pack_weight_kernel(const float* __restrict__ W, __bf16* __restrict__ P,
                                   int K, int Nc) {
  int KT = (K + 31) >> 5;
  int tile = blockIdx.x;            // ntile*KT + ktile
  int lane = threadIdx.x & 31;
  int ktile = tile % KT;
  int ntile = tile / KT;
  int n = ntile * 16 + (lane & 15);
  int kbase = ktile * 32 + ((lane >> 4) ? 8 : 0);
  bf16x16 v;
#pragma unroll
  for (int j = 0; j < 16; ++j) {
    int k = kbase + j + ((j >= 8) ? 8 : 0);
    float f = (k < K && n < Nc) ? W[(size_t)k * Nc + n] : 0.0f;
    v[j] = (__bf16)f;
  }
  *(bf16x16*)(P + ((size_t)tile * 32 + lane) * 16) = v;
}

// ---------------------------------------------------------------------------
// Generic WMMA GEMM: C[M,Nc] = act(A[M,K] @ W + bias) (+ C if ACCUM)
// Requires M%128==0, K%32==0, Nc%64==0 (true for every launch here).
// block tile 128x64, 8 waves in 4x2, wave tile 32x32 (2x2 fragments).
// A tile staged fp32 into LDS via gfx1250 async global->LDS copies when
// available (ASYNCcnt path); bf16 conversion happens at fragment load.
// ACT: 0=none 1=relu 2=silu
// ---------------------------------------------------------------------------
template <int ACT, int ACCUM>
__global__ __launch_bounds__(256) void gemm_wmma_kernel(
    const float* __restrict__ A, const __bf16* __restrict__ Bp,
    const float* __restrict__ bias, float* __restrict__ Cout,
    int M, int K, int Nc) {
  __shared__ float Asf[128][32];
  const int KT = K >> 5;
  const int m0 = blockIdx.x * 128;
  const int n0 = blockIdx.y * 64;
  const int tid = threadIdx.x;
  const int wave = tid >> 5, lane = tid & 31;
  const int wm = wave >> 1, wn = wave & 1;
  const int lrow = lane & 15;
  const int hi = lane >> 4;
  const f32x8 zero8 = {0.f,0.f,0.f,0.f,0.f,0.f,0.f,0.f};
  f32x8 acc[2][2] = {{zero8, zero8}, {zero8, zero8}};

  for (int kt = 0; kt < KT; ++kt) {
    __syncthreads();
    // stage A tile (128x32 fp32): 1024 x float4, 4 per thread
#pragma unroll
    for (int i = 0; i < 4; ++i) {
      int id4 = tid + i * 256;
      int r = id4 >> 3;
      int c = (id4 & 7) * 4;
      const float* gp = A + (size_t)(m0 + r) * K + kt * 32 + c;
#if USE_ASYNC_LDS
      __builtin_amdgcn_global_load_async_to_lds_b128(
          (as1_v4i*)gp, (as3_v4i*)&Asf[r][c], 0, 0);
#else
      *(float4*)&Asf[r][c] = *(const float4*)gp;
#endif
    }
    if (kt + 1 < KT) {
      // prefetch next K tile (global_prefetch_b8)
      __builtin_prefetch(A + (size_t)(m0 + (tid >> 1)) * K + (kt + 1) * 32 + (tid & 1) * 16, 0, 0);
    }
#if USE_ASYNC_LDS
    __builtin_amdgcn_s_wait_asynccnt(0);
#endif
    __syncthreads();
    // A fragments from LDS (fp32 -> bf16)
    bf16x16 af[2];
#pragma unroll
    for (int fm = 0; fm < 2; ++fm) {
      const float* ap = &Asf[wm*32 + fm*16 + lrow][hi ? 8 : 0];
#pragma unroll
      for (int j = 0; j < 8; ++j) { af[fm][j] = (__bf16)ap[j]; af[fm][j+8] = (__bf16)ap[j+16]; }
    }
    // B fragments straight from packed global + WMMA
#pragma unroll
    for (int fn = 0; fn < 2; ++fn) {
      int ntile = (n0 >> 4) + wn*2 + fn;
      bf16x16 bfv = *(const bf16x16*)(Bp + ((size_t)(ntile * KT + kt) * 32 + lane) * 16);
#pragma unroll
      for (int fm = 0; fm < 2; ++fm)
        acc[fm][fn] = wmma_bf16(af[fm], bfv, acc[fm][fn]);
    }
  }
  // epilogue: C layout - lane<16: N=n, M=r ; lane>=16: N=lane-16, M=8+r
#pragma unroll
  for (int fm = 0; fm < 2; ++fm) {
#pragma unroll
    for (int fn = 0; fn < 2; ++fn) {
      int ncol = n0 + wn*32 + fn*16 + lrow;
      int mb = m0 + wm*32 + fm*16 + (hi ? 8 : 0);
      float bv = bias ? bias[ncol] : 0.f;
#pragma unroll
      for (int r = 0; r < 8; ++r) {
        int m = mb + r;
        float v = acc[fm][fn][r] + bv;
        if (ACT == 1) v = fmaxf(v, 0.f);
        else if (ACT == 2) v = v / (1.f + __expf(-v));
        float* cp = Cout + (size_t)m * Nc + ncol;
        if (ACCUM) v += *cp;
        *cp = v;
      }
    }
  }
}

// ---------------------------------------------------------------------------
// Edge projection GEMM fused with ReLU + atomic scatter-add by src node:
// ef_node[src[e]] += relu(edge_attr[e] @ ep_w + ep_b).  K=16 padded to 32.
// ---------------------------------------------------------------------------
__global__ __launch_bounds__(256) void ep_gemm_scatter_kernel(
    const float* __restrict__ EA, const __bf16* __restrict__ Bp,
    const float* __restrict__ bias, const int* __restrict__ src,
    float* __restrict__ efout) {
  __shared__ __bf16 As[128][32];
  __shared__ int srcs[128];
  const int m0 = blockIdx.x * 128;
  const int n0 = blockIdx.y * 64;
  const int tid = threadIdx.x;
  const int wave = tid >> 5, lane = tid & 31;
  const int wm = wave >> 1, wn = wave & 1;
  const int lrow = lane & 15, hi = lane >> 4;
#pragma unroll
  for (int i = 0; i < 4; ++i) {
    int idx = tid + i * 256;
    int r = idx >> 3;
    int c = (idx & 7) * 4;
    float4 v = make_float4(0.f, 0.f, 0.f, 0.f);
    if (c < EF_DIM) v = *(const float4*)(EA + (size_t)(m0 + r) * EF_DIM + c);
    As[r][c+0] = (__bf16)v.x; As[r][c+1] = (__bf16)v.y;
    As[r][c+2] = (__bf16)v.z; As[r][c+3] = (__bf16)v.w;
  }
  if (tid < 128) srcs[tid] = src[m0 + tid];
  __syncthreads();
  const f32x8 zero8 = {0.f,0.f,0.f,0.f,0.f,0.f,0.f,0.f};
  f32x8 acc[2][2] = {{zero8, zero8}, {zero8, zero8}};
  bf16x16 af[2];
#pragma unroll
  for (int fm = 0; fm < 2; ++fm) {
    const __bf16* ap = &As[wm*32 + fm*16 + lrow][hi ? 8 : 0];
    bf16x8 lo = *(const bf16x8*)ap;
    bf16x8 hh = *(const bf16x8*)(ap + 16);
#pragma unroll
    for (int j = 0; j < 8; ++j) { af[fm][j] = lo[j]; af[fm][j+8] = hh[j]; }
  }
#pragma unroll
  for (int fn = 0; fn < 2; ++fn) {
    int ntile = (n0 >> 4) + wn*2 + fn;          // KT == 1
    bf16x16 bfv = *(const bf16x16*)(Bp + ((size_t)ntile * 32 + lane) * 16);
#pragma unroll
    for (int fm = 0; fm < 2; ++fm)
      acc[fm][fn] = wmma_bf16(af[fm], bfv, acc[fm][fn]);
  }
#pragma unroll
  for (int fm = 0; fm < 2; ++fm) {
#pragma unroll
    for (int fn = 0; fn < 2; ++fn) {
      int ncol = n0 + wn*32 + fn*16 + lrow;
      int mloc = wm*32 + fm*16 + (hi ? 8 : 0);
      float bv = bias[ncol];
#pragma unroll
      for (int r = 0; r < 8; ++r) {
        float v = fmaxf(acc[fm][fn][r] + bv, 0.f);
        int s = srcs[mloc + r];
        atomicAdd(&efout[(size_t)s * C_DIM + ncol], v);
      }
    }
  }
}

// ---------------------------------------------------------------------------
// degree / edge-weight / dinv
// ---------------------------------------------------------------------------
__global__ void deg_init_kernel(float* deg) {
  int i = blockIdx.x * 256 + threadIdx.x;
  if (i < N_NODES) deg[i] = 1.0f;          // self-loop weight
}
__global__ void edge_ew_kernel(const float* __restrict__ ea, const int* __restrict__ dst,
                               float* __restrict__ ew, float* __restrict__ deg) {
  int e = blockIdx.x * 256 + threadIdx.x;
  if (e >= E_EDGES) return;
  const float4* a = (const float4*)(ea + (size_t)e * EF_DIM);
  float s = 0.f;
#pragma unroll
  for (int i = 0; i < 4; ++i) { float4 v = a[i]; s += v.x*v.x + v.y*v.y + v.z*v.z + v.w*v.w; }
  float w = sqrtf(s);
  ew[e] = w;
  atomicAdd(&deg[dst[e]], w);
}
__global__ void dinv_kernel(const float* __restrict__ deg, float* __restrict__ dinv) {
  int i = blockIdx.x * 256 + threadIdx.x;
  if (i >= N_NODES) return;
  float d = deg[i];
  dinv[i] = d > 0.f ? rsqrtf(fmaxf(d, 1e-12f)) : 0.f;
}

// gather xw[src], scale by sym-norm, atomic scatter into x_conv[dst]; 1 wave/edge
__global__ __launch_bounds__(256) void gcn_scatter_kernel(
    const float* __restrict__ xw, const int* __restrict__ src, const int* __restrict__ dst,
    const float* __restrict__ ew, const float* __restrict__ dinv, float* __restrict__ xconv) {
  int e = blockIdx.x * 8 + (threadIdx.x >> 5);
  int lane = threadIdx.x & 31;
  if (e >= E_EDGES) return;
  int s = src[e], d = dst[e];
  float nrm = dinv[s] * ew[e] * dinv[d];
  const float4* xr = (const float4*)(xw + (size_t)s * C_DIM);
  float* outp = xconv + (size_t)d * C_DIM;
#pragma unroll
  for (int i = 0; i < 4; ++i) {
    float4 v = xr[lane + i * 32];
    int c = (lane + i * 32) * 4;
    atomicAdd(outp + c + 0, v.x * nrm);
    atomicAdd(outp + c + 1, v.y * nrm);
    atomicAdd(outp + c + 2, v.z * nrm);
    atomicAdd(outp + c + 3, v.w * nrm);
  }
}

// self-loop contribution (norm = dinv[i]^2) + gcn bias (non-atomic; serialized)
__global__ void selfloop_bias_kernel(const float* __restrict__ xw, const float* __restrict__ dinv,
                                     const float* __restrict__ gcn_b, float* __restrict__ xconv) {
  int idx = blockIdx.x * 256 + threadIdx.x;       // one float4 each
  int row = idx >> 7;
  int c = (idx & 127) * 4;
  if (row >= N_NODES) return;
  float s = dinv[row]; s = s * s;
  float4 xv = *(const float4*)(xw + (size_t)row * C_DIM + c);
  float4 bv = *(const float4*)(gcn_b + c);
  float4* o = (float4*)(xconv + (size_t)row * C_DIM + c);
  float4 cur = *o;
  cur.x += s * xv.x + bv.x; cur.y += s * xv.y + bv.y;
  cur.z += s * xv.z + bv.z; cur.w += s * xv.w + bv.w;
  *o = cur;
}

// h = sig(g)*xc + (1-sig(g))*ef ; out = relu(LN(h)) + xin   (1 wave per row)
__global__ __launch_bounds__(256) void gate_ln_kernel(
    const float* __restrict__ g, const float* __restrict__ xc, const float* __restrict__ ef,
    const float* __restrict__ xin, const float* __restrict__ gamma, const float* __restrict__ beta,
    float* __restrict__ outp) {
  int row = blockIdx.x * 8 + (threadIdx.x >> 5);
  int lane = threadIdx.x & 31;
  size_t base = (size_t)row * C_DIM;
  float h[16];
  float s = 0.f, ss = 0.f;
#pragma unroll
  for (int i = 0; i < 4; ++i) {
    int c = lane * 16 + i * 4;
    float4 gv = *(const float4*)(g + base + c);
    float4 xv = *(const float4*)(xc + base + c);
    float4 ev = *(const float4*)(ef + base + c);
    float gg[4] = {gv.x, gv.y, gv.z, gv.w};
    float xx[4] = {xv.x, xv.y, xv.z, xv.w};
    float ee[4] = {ev.x, ev.y, ev.z, ev.w};
#pragma unroll
    for (int j = 0; j < 4; ++j) {
      float sg = 1.f / (1.f + __expf(-gg[j]));
      float hv = sg * xx[j] + (1.f - sg) * ee[j];
      h[i*4+j] = hv; s += hv; ss += hv * hv;
    }
  }
  s = wave_sum32(s); ss = wave_sum32(ss);
  float mean = s * (1.f / 512.f);
  float var = ss * (1.f / 512.f) - mean * mean;
  float inv = rsqrtf(var + 1e-5f);
#pragma unroll
  for (int i = 0; i < 16; ++i) {
    int c = lane * 16 + i;
    float o = (h[i] - mean) * inv * gamma[c] + beta[c];
    o = fmaxf(o, 0.f);
    outp[base + c] = o + xin[base + c];
  }
}

// out = LN(a + b) (1 wave per row)
__global__ __launch_bounds__(256) void ln_residual_kernel(
    const float* __restrict__ a, const float* __restrict__ b,
    const float* __restrict__ gamma, const float* __restrict__ beta,
    float* __restrict__ outp) {
  int row = blockIdx.x * 8 + (threadIdx.x >> 5);
  int lane = threadIdx.x & 31;
  size_t base = (size_t)row * C_DIM;
  float h[16];
  float s = 0.f, ss = 0.f;
#pragma unroll
  for (int i = 0; i < 4; ++i) {
    int c = lane * 16 + i * 4;
    float4 av = *(const float4*)(a + base + c);
    float4 bv = *(const float4*)(b + base + c);
    float t[4] = {av.x + bv.x, av.y + bv.y, av.z + bv.z, av.w + bv.w};
#pragma unroll
    for (int j = 0; j < 4; ++j) { h[i*4+j] = t[j]; s += t[j]; ss += t[j] * t[j]; }
  }
  s = wave_sum32(s); ss = wave_sum32(ss);
  float mean = s * (1.f / 512.f);
  float var = ss * (1.f / 512.f) - mean * mean;
  float inv = rsqrtf(var + 1e-5f);
#pragma unroll
  for (int i = 0; i < 16; ++i) {
    int c = lane * 16 + i;
    outp[base + c] = (h[i] - mean) * inv * gamma[c] + beta[c];
  }
}

// ---------------------------------------------------------------------------
// Attention: exact 256x256 per (graph, head); graphs are contiguous 256-node
// runs so no dense-batch padding / masking is needed.
// One block per (b, h, 32-row query tile). 8 waves.
// ---------------------------------------------------------------------------
__global__ __launch_bounds__(256) void attn_kernel(const float* __restrict__ qkv,
                                                   float* __restrict__ obuf) {
  __shared__ float  S[32][256];      // score / prob tile (fp32)
  __shared__ __bf16 VT[64][256];     // V transposed (Dh x L) bf16
  int blk = blockIdx.x;
  int qt = blk & 7;
  int h  = (blk >> 3) & 7;
  int b  = blk >> 6;
  int base = b * NPG;
  const size_t ldq = 1536;
  int qoff = h * 64;
  int koff = 512 + h * 64;
  int voff = 1024 + h * 64;
  int tid = threadIdx.x, wave = tid >> 5, lane = tid & 31;
  int lrow = lane & 15, hi = lane >> 4;

  // stage V^T
  for (int i = tid; i < 256 * 64; i += 256) {
    int pos = i >> 6, d = i & 63;
    VT[d][pos] = (__bf16)qkv[(size_t)(base + pos) * ldq + voff + d];
  }

  // S = (Q * 1/sqrt(Dh)) @ K^T : wave -> (mt = wave&1) x 64-col group (wave>>1)
  int mt = wave & 1;
  int ng = wave >> 1;
  const f32x8 zero8 = {0.f,0.f,0.f,0.f,0.f,0.f,0.f,0.f};
  f32x8 sacc[4] = {zero8, zero8, zero8, zero8};
#pragma unroll
  for (int ks = 0; ks < 2; ++ks) {                 // Dh=64, K-step 32
    int kb = ks * 32 + (hi ? 8 : 0);
    const float* qp = qkv + (size_t)(base + qt*32 + mt*16 + lrow) * ldq + qoff + kb;
    bf16x16 af;
#pragma unroll
    for (int j = 0; j < 8; ++j) {
      af[j]   = (__bf16)(qp[j]      * 0.125f);
      af[j+8] = (__bf16)(qp[j + 16] * 0.125f);
    }
#pragma unroll
    for (int nt = 0; nt < 4; ++nt) {
      const float* kp = qkv + (size_t)(base + ng*64 + nt*16 + lrow) * ldq + koff + kb;
      bf16x16 bfv;
#pragma unroll
      for (int j = 0; j < 8; ++j) { bfv[j] = (__bf16)kp[j]; bfv[j+8] = (__bf16)kp[j+16]; }
      sacc[nt] = wmma_bf16(af, bfv, sacc[nt]);
    }
  }
#pragma unroll
  for (int nt = 0; nt < 4; ++nt) {
    int col = ng*64 + nt*16 + lrow;
    int mr = mt*16 + (hi ? 8 : 0);
#pragma unroll
    for (int r = 0; r < 8; ++r) S[mr + r][col] = sacc[nt][r];
  }
  __syncthreads();

  // row softmax: 32 rows, 4 per wave
#pragma unroll
  for (int rr = 0; rr < 4; ++rr) {
    int r = wave * 4 + rr;
    float v[8];
    float mx = -1e30f;
#pragma unroll
    for (int i = 0; i < 8; ++i) { v[i] = S[r][lane*8 + i]; mx = fmaxf(mx, v[i]); }
    mx = wave_max32(mx);
    float sm = 0.f;
#pragma unroll
    for (int i = 0; i < 8; ++i) { v[i] = __expf(v[i] - mx); sm += v[i]; }
    sm = wave_sum32(sm);
    float inv = 1.f / sm;
#pragma unroll
    for (int i = 0; i < 8; ++i) S[r][lane*8 + i] = v[i] * inv;
  }
  __syncthreads();

  // O = P @ V : one 16x16 fragment per wave (2x4 grid), K loop 256/32
  int mt2 = wave >> 2, nt2 = wave & 3;
  f32x8 oacc = zero8;
#pragma unroll
  for (int ks = 0; ks < 8; ++ks) {
    int kb = ks * 32 + (hi ? 8 : 0);
    const float* ap = &S[mt2*16 + lrow][kb];
    bf16x16 af;
#pragma unroll
    for (int j = 0; j < 8; ++j) { af[j] = (__bf16)ap[j]; af[j+8] = (__bf16)ap[j+16]; }
    const __bf16* vp = &VT[nt2*16 + lrow][kb];
    bf16x16 bfv;
#pragma unroll
    for (int j = 0; j < 8; ++j) { bfv[j] = vp[j]; bfv[j+8] = vp[j+16]; }
    oacc = wmma_bf16(af, bfv, oacc);
  }
  int orow = base + qt*32 + mt2*16 + (hi ? 8 : 0);
  int ocol = h*64 + nt2*16 + lrow;
#pragma unroll
  for (int r = 0; r < 8; ++r) obuf[(size_t)(orow + r) * C_DIM + ocol] = oacc[r];
}

// ---------------------------------------------------------------------------
extern "C" void kernel_launch(void* const* d_in, const int* in_sizes, int n_in,
                              void* d_out, int out_size, void* d_ws, size_t ws_size,
                              hipStream_t stream) {
  (void)in_sizes; (void)n_in; (void)out_size; (void)ws_size;
  const float* x         = (const float*)d_in[0];
  const float* edge_attr = (const float*)d_in[1];
  const float* gcn_w     = (const float*)d_in[2];
  const float* gcn_b     = (const float*)d_in[3];
  const float* ep_w      = (const float*)d_in[4];
  const float* ep_b      = (const float*)d_in[5];
  const float* gate_w    = (const float*)d_in[6];
  const float* gate_b    = (const float*)d_in[7];
  const float* n1_g      = (const float*)d_in[8];
  const float* n1_b      = (const float*)d_in[9];
  const float* in_w      = (const float*)d_in[10];
  const float* in_b      = (const float*)d_in[11];
  const float* out_w     = (const float*)d_in[12];
  const float* out_b     = (const float*)d_in[13];
  const float* tn_g      = (const float*)d_in[14];
  const float* tn_b      = (const float*)d_in[15];
  const float* m_w1      = (const float*)d_in[16];
  const float* m_b1      = (const float*)d_in[17];
  const float* m_w2      = (const float*)d_in[18];
  const float* m_b2      = (const float*)d_in[19];
  const float* fn_g      = (const float*)d_in[20];
  const float* fn_b      = (const float*)d_in[21];
  const int*   src       = (const int*)d_in[22];
  const int*   dst       = ((const int*)d_in[22]) + E_EDGES;

  const size_t NC = (size_t)N_NODES * C_DIM;
  char* wsp = (char*)d_ws;
  size_t off = 0;
  auto take = [&](size_t bytes) -> char* {
    char* p = wsp + off;
    off += (bytes + 255) & ~(size_t)255;
    return p;
  };
  float*  ew    = (float*)take((size_t)E_EDGES * 4);
  float*  deg   = (float*)take((size_t)N_NODES * 4);
  float*  dinv  = (float*)take((size_t)N_NODES * 4);
  float*  xw    = (float*)take(NC * 4);                 // later aliased: attn_out
  float*  xconv = (float*)take(NC * 4);                 // later aliased: obuf
  float*  efn   = (float*)take(NC * 4);                 // later aliased: mlp_out
  float*  gate  = (float*)take(NC * 4);
  float*  xres  = (float*)take(NC * 4);
  float*  qkv   = (float*)take((size_t)N_NODES * 1536 * 4); // later aliased: mlp_h
  float*  x2    = (float*)take(NC * 4);
  __bf16* p_gcn = (__bf16*)take((size_t)512 * 512 * 2);
  __bf16* p_ep  = (__bf16*)take((size_t)32  * 512 * 2);
  __bf16* p_gt  = (__bf16*)take((size_t)512 * 512 * 2);
  __bf16* p_gb  = (__bf16*)take((size_t)512 * 512 * 2);
  __bf16* p_in  = (__bf16*)take((size_t)512 * 1536 * 2);
  __bf16* p_out = (__bf16*)take((size_t)512 * 512 * 2);
  __bf16* p_m1  = (__bf16*)take((size_t)512 * 1024 * 2);
  __bf16* p_m2  = (__bf16*)take((size_t)1024 * 512 * 2);
  float* attn_out = xw;     // xw dead after conv
  float* obuf     = xconv;  // xconv dead after gate_ln
  float* mlp_h    = qkv;    // qkv dead after attention
  float* mlp_out  = efn;    // efn dead after gate_ln

  // ---- pack all weights into WMMA-fragment-native bf16 layout ----
  pack_weight_kernel<<<dim3(32 * 16), 32, 0, stream>>>(gcn_w, p_gcn, 512, 512);
  pack_weight_kernel<<<dim3(32 * 1),  32, 0, stream>>>(ep_w,  p_ep,  16,  512);
  pack_weight_kernel<<<dim3(32 * 16), 32, 0, stream>>>(gate_w,             p_gt, 512, 512);
  pack_weight_kernel<<<dim3(32 * 16), 32, 0, stream>>>(gate_w + 512 * 512, p_gb, 512, 512);
  pack_weight_kernel<<<dim3(96 * 16), 32, 0, stream>>>(in_w,  p_in,  512, 1536);
  pack_weight_kernel<<<dim3(32 * 16), 32, 0, stream>>>(out_w, p_out, 512, 512);
  pack_weight_kernel<<<dim3(64 * 16), 32, 0, stream>>>(m_w1,  p_m1,  512, 1024);
  pack_weight_kernel<<<dim3(32 * 32), 32, 0, stream>>>(m_w2,  p_m2,  1024, 512);

  // ---- GCN conv ----
  (void)hipMemsetAsync(xconv, 0, NC * 4, stream);
  (void)hipMemsetAsync(efn,   0, NC * 4, stream);
  deg_init_kernel<<<N_NODES / 256, 256, 0, stream>>>(deg);
  edge_ew_kernel<<<E_EDGES / 256, 256, 0, stream>>>(edge_attr, dst, ew, deg);
  dinv_kernel<<<N_NODES / 256, 256, 0, stream>>>(deg, dinv);
  gemm_wmma_kernel<0, 0><<<dim3(N_NODES / 128, 512 / 64), 256, 0, stream>>>(
      x, p_gcn, nullptr, xw, N_NODES, 512, 512);
  gcn_scatter_kernel<<<E_EDGES / 8, 256, 0, stream>>>(xw, src, dst, ew, dinv, xconv);
  selfloop_bias_kernel<<<(N_NODES * 128) / 256, 256, 0, stream>>>(xw, dinv, gcn_b, xconv);

  // ---- edge proj + scatter ----
  ep_gemm_scatter_kernel<<<dim3(E_EDGES / 128, 512 / 64), 256, 0, stream>>>(
      edge_attr, p_ep, ep_b, src, efn);

  // ---- gate (split-K concat) + LN + residual ----
  gemm_wmma_kernel<0, 0><<<dim3(N_NODES / 128, 512 / 64), 256, 0, stream>>>(
      xconv, p_gt, gate_b, gate, N_NODES, 512, 512);
  gemm_wmma_kernel<0, 1><<<dim3(N_NODES / 128, 512 / 64), 256, 0, stream>>>(
      efn, p_gb, nullptr, gate, N_NODES, 512, 512);
  gate_ln_kernel<<<N_NODES / 8, 256, 0, stream>>>(gate, xconv, efn, x, n1_g, n1_b, xres);

  // ---- MHA ----
  gemm_wmma_kernel<0, 0><<<dim3(N_NODES / 128, 1536 / 64), 256, 0, stream>>>(
      xres, p_in, in_b, qkv, N_NODES, 512, 1536);
  attn_kernel<<<B_GR * H_HEADS * 8, 256, 0, stream>>>(qkv, obuf);
  gemm_wmma_kernel<0, 0><<<dim3(N_NODES / 128, 512 / 64), 256, 0, stream>>>(
      obuf, p_out, out_b, attn_out, N_NODES, 512, 512);
  ln_residual_kernel<<<N_NODES / 8, 256, 0, stream>>>(xres, attn_out, tn_g, tn_b, x2);

  // ---- MLP + final LN ----
  gemm_wmma_kernel<2, 0><<<dim3(N_NODES / 128, 1024 / 64), 256, 0, stream>>>(
      x2, p_m1, m_b1, mlp_h, N_NODES, 512, 1024);
  gemm_wmma_kernel<0, 0><<<dim3(N_NODES / 128, 512 / 64), 256, 0, stream>>>(
      mlp_h, p_m2, m_b2, mlp_out, N_NODES, 1024, 512);
  ln_residual_kernel<<<N_NODES / 8, 256, 0, stream>>>(x2, mlp_out, fn_g, fn_b, (float*)d_out);
}
